// GatingNetwork_3453153706031
// MI455X (gfx1250) — matrix operand order repaired and checked by reading, not compiled, not run on hardware
//
#include <hip/hip_runtime.h>

// ---------------------------------------------------------------------------
// PLE/CGC multi-gate MoE for MI455X (gfx1250, wave32, WMMA bf16).
//   B=32768, I=384, U=128, E=8, G=256.
// Dominant cost: 3 x (B x I x (E*U)) expert GEMMs = 77.3 GFLOP -> run via
// v_wmma_f32_16x16x32_bf16 (bf16 A/B, fp32 accum). Min HBM traffic ~300MB
// (~13us @ 23.3 TB/s); X-tile reuse across the 8 experts served by 192MB L2
// via block ordering (tile*8 + e).
// Workspace layout: [0, 8MB) combined gate coeffs [B][8][8] f32,
//                   [8MB, 8MB+2.36MB) bf16 expert weights [3][E][U][I].
// ---------------------------------------------------------------------------

#define B_ 32768
#define I_ 384
#define U_ 128
#define E_ 8
#define G_ 256
#define ALPHA_ 0.1f
#define BETA_ 0.1f

typedef __attribute__((ext_vector_type(16))) __bf16 v16bf;
typedef __attribute__((ext_vector_type(8)))  float  v8f;

struct U8x32 { uint4 lo, hi; };

__device__ inline unsigned short f2bf(float f) {
  unsigned u = __builtin_bit_cast(unsigned, f);
  u += 0x7FFFu + ((u >> 16) & 1u);          // round-to-nearest-even
  return (unsigned short)(u >> 16);
}

__device__ inline v16bf load_frag(const unsigned short* b0, const unsigned short* b1) {
  U8x32 tmp;
  tmp.lo = *(const uint4*)b0;               // ds_load_b128
  tmp.hi = *(const uint4*)b1;               // ds_load_b128
  return __builtin_bit_cast(v16bf, tmp);
}

// ---------------------------------------------------------------------------
// Kernel 0: fp32 -> bf16 weight conversion
// ---------------------------------------------------------------------------
__global__ void convert_w_kernel(const float* __restrict__ src,
                                 unsigned short* __restrict__ dst, int n) {
  int i = blockIdx.x * 256 + threadIdx.x;
  if (i < n) dst[i] = f2bf(src[i]);
}

// ---------------------------------------------------------------------------
// Kernel 1: gates + algebraic combination into 8 coefficient vectors per row.
//   cat0 = g_t1[0:8]  + a*g_s1_ip   (sh  -> out1)
//   cat1 = g_t1[8:16]               (t1  -> out1)
//   cat2 = a*(g_s1_up + g_t1_ui)    (t2  -> out1)
//   cat3 = g_t2[0:8]  + b*g_share2  (sh  -> out2)
//   cat4 = g_t2[8:16] + b*(g_t2_up + g_t2_ip)   (t2 -> out2)
//   cat5 = g_shared[0:8]            (t1  -> out_sh)
//   cat6 = g_shared[8:16]           (sh  -> out_sh)
//   cat7 = g_shared[16:24]          (t2  -> out_sh)
// ---------------------------------------------------------------------------
template<int NOUT>
__device__ inline void gate_dots(const float* __restrict__ xrow,
                                 const float* __restrict__ W,
                                 const float* __restrict__ bias,
                                 int n_in, float* out) {
  float acc[NOUT];
  #pragma unroll
  for (int o = 0; o < NOUT; ++o) acc[o] = bias[o];
  for (int k0 = 0; k0 < n_in; k0 += 8) {
    float xv[8];
    #pragma unroll
    for (int j = 0; j < 8; ++j) xv[j] = xrow[k0 + j];
    #pragma unroll
    for (int o = 0; o < NOUT; ++o) {
      const float* w = W + o * n_in + k0;    // thread-uniform -> s_load
      #pragma unroll
      for (int j = 0; j < 8; ++j) acc[o] = fmaf(xv[j], w[j], acc[o]);
    }
  }
  #pragma unroll
  for (int o = 0; o < NOUT; ++o) out[o] = acc[o];
}

template<int N>
__device__ inline void softmaxN(float* v) {
  float m = v[0];
  #pragma unroll
  for (int i = 1; i < N; ++i) m = fmaxf(m, v[i]);
  float s = 0.f;
  #pragma unroll
  for (int i = 0; i < N; ++i) { v[i] = __expf(v[i] - m); s += v[i]; }
  float inv = 1.f / s;
  #pragma unroll
  for (int i = 0; i < N; ++i) v[i] *= inv;
}

__global__ __launch_bounds__(256)
void gates_kernel(const float* __restrict__ xsh, const float* __restrict__ xt1,
                  const float* __restrict__ xt2, const float* __restrict__ uikey,
                  const float* __restrict__ upkey, const float* __restrict__ ipkey,
                  const float* Wg_shared, const float* bg_shared,
                  const float* Wg_t1, const float* bg_t1,
                  const float* Wg_t2, const float* bg_t2,
                  const float* Wg_share2, const float* bg_share2,
                  const float* Wg_t2_ip, const float* bg_t2_ip,
                  const float* Wg_t2_up, const float* bg_t2_up,
                  const float* Wg_s1_ip, const float* bg_s1_ip,
                  const float* Wg_s1_up, const float* bg_s1_up,
                  const float* Wg_t1_ui, const float* bg_t1_ui,
                  float* __restrict__ C) {
  int b = blockIdx.x * 256 + threadIdx.x;
  if (b >= B_) return;
  float* crow = C + (size_t)b * 64;

  {   // shared-tower gate (24-way), pure copies into cats 5/6/7
    float g[24];
    gate_dots<24>(xsh + (size_t)b * I_, Wg_shared, bg_shared, I_, g);
    softmaxN<24>(g);
    #pragma unroll
    for (int j = 0; j < 8; ++j) {
      crow[40 + j] = g[j]; crow[48 + j] = g[8 + j]; crow[56 + j] = g[16 + j];
    }
  }
  {   // task1 side
    float gt1[16], gip[8], gup[8], gui[8];
    gate_dots<16>(xt1 + (size_t)b * I_, Wg_t1, bg_t1, I_, gt1);     softmaxN<16>(gt1);
    gate_dots<8>(ipkey + (size_t)b * G_, Wg_s1_ip, bg_s1_ip, G_, gip); softmaxN<8>(gip);
    gate_dots<8>(upkey + (size_t)b * G_, Wg_s1_up, bg_s1_up, G_, gup); softmaxN<8>(gup);
    gate_dots<8>(uikey + (size_t)b * G_, Wg_t1_ui, bg_t1_ui, G_, gui); softmaxN<8>(gui);
    #pragma unroll
    for (int j = 0; j < 8; ++j) {
      crow[0 + j]  = gt1[j] + ALPHA_ * gip[j];
      crow[8 + j]  = gt1[8 + j];
      crow[16 + j] = ALPHA_ * (gup[j] + gui[j]);
    }
  }
  {   // task2 side
    float gt2[16], gs2[8], gu[8], gi[8];
    gate_dots<16>(xt2 + (size_t)b * I_, Wg_t2, bg_t2, I_, gt2);      softmaxN<16>(gt2);
    gate_dots<8>(uikey + (size_t)b * G_, Wg_share2, bg_share2, G_, gs2); softmaxN<8>(gs2);
    gate_dots<8>(upkey + (size_t)b * G_, Wg_t2_up, bg_t2_up, G_, gu);    softmaxN<8>(gu);
    gate_dots<8>(ipkey + (size_t)b * G_, Wg_t2_ip, bg_t2_ip, G_, gi);    softmaxN<8>(gi);
    #pragma unroll
    for (int j = 0; j < 8; ++j) {
      crow[24 + j] = gt2[j] + BETA_ * gs2[j];
      crow[32 + j] = gt2[8 + j] + BETA_ * (gu[j] + gi[j]);
    }
  }
}

// ---------------------------------------------------------------------------
// Kernel 2: expert GEMMs (bf16 WMMA) + gated mix.
// Block (tile, e): input rows [128*tile, +128), expert e; owns output rows
// b' = 4096*e + 16*tile + p (p=0..15) of all three outputs exclusively.
// 8 waves: wave w -> M-rows 32*(w&3), N-cols 64*(w>>2); 2x4 C-tiles each.
// ---------------------------------------------------------------------------
__global__ __launch_bounds__(256)
void expert_mix_kernel(const float* __restrict__ xsh,
                       const float* __restrict__ xt1,
                       const float* __restrict__ xt2,
                       const unsigned short* __restrict__ wbf,   // [3][E][U][I] bf16
                       const float* __restrict__ bsh,
                       const float* __restrict__ bt1,
                       const float* __restrict__ bt2,
                       const float* __restrict__ Cws,            // [B][8][8]
                       float* __restrict__ out) {                // [3][B][U]
  constexpr int KC = 96, KCP = 104, YP = 132;
  __shared__ __align__(16) unsigned char smem[128 * YP * 4];     // 67584 B
  unsigned short* sA = (unsigned short*)smem;                    // X chunk 128xKCP bf16
  unsigned short* sB = (unsigned short*)(smem + 128 * KCP * 2);  // W chunk 128xKCP bf16
  float* sY = (float*)smem;                                      // Y tile (time-disjoint alias)

  const int tile = blockIdx.x >> 3;
  const int e    = blockIdx.x & 7;
  const int t    = threadIdx.x;
  const int wave = t >> 5;
  const int lane = t & 31;
  const int lmod = lane & 15, lhalf = lane >> 4;
  const int mblk = (wave & 3) * 32;
  const int nblk = (wave >> 2) * 64;
  const int p  = t >> 4;             // output row in tile (0..15)
  const int u0 = (t & 15) * 8;       // output unit base
  const int bprime = (e << 12) + (tile << 4) + p;

  float acc_sh[8], acc1[8], acc2[8];
  #pragma unroll
  for (int q = 0; q < 8; ++q) acc_sh[q] = acc1[q] = acc2[q] = 0.f;

  const float* xs[3] = {xsh, xt1, xt2};
  const float* bs[3] = {bsh, bt1, bt2};
  const float* crow  = Cws + (size_t)bprime * 64;

  #pragma unroll
  for (int tw = 0; tw < 3; ++tw) {
    const float* X = xs[tw] + (size_t)tile * 128 * I_;
    const unsigned short* W = wbf + ((size_t)tw * E_ + e) * (U_ * I_);

    v8f vzero = {0.f, 0.f, 0.f, 0.f, 0.f, 0.f, 0.f, 0.f};
    v8f acc[2][4];
    #pragma unroll
    for (int mi = 0; mi < 2; ++mi)
      #pragma unroll
      for (int ni = 0; ni < 4; ++ni) acc[mi][ni] = vzero;

    for (int kc = 0; kc < I_; kc += KC) {
      // stage X chunk fp32->bf16 (6144 bf16-pairs, coalesced float2 loads)
      for (int i = t; i < 6144; i += 256) {
        int row = i / 48, c = i % 48;
        float2 v = *(const float2*)(X + row * I_ + kc + c * 2);
        unsigned pk = (unsigned)f2bf(v.x) | ((unsigned)f2bf(v.y) << 16);
        ((unsigned*)sA)[row * (KCP / 2) + c] = pk;
      }
      // stage W chunk (already bf16; dword copies)
      for (int i = t; i < 6144; i += 256) {
        int row = i / 48, c = i % 48;
        ((unsigned*)sB)[row * (KCP / 2) + c] =
            ((const unsigned*)W)[row * (I_ / 2) + kc / 2 + c];
      }
      __syncthreads();

      #pragma unroll
      for (int ks = 0; ks < 3; ++ks) {
        const int kk = ks * 32;
        v16bf afrag[2];
        #pragma unroll
        for (int mi = 0; mi < 2; ++mi) {
          // A 16x32 bf16 layout: lane half h holds K = {8h..8h+7, 16+8h..16+8h+7}
          const unsigned short* r = sA + (mblk + 16 * mi + lmod) * KCP;
          afrag[mi] = load_frag(r + kk + 8 * lhalf, r + kk + 16 + 8 * lhalf);
        }
        #pragma unroll
        for (int ni = 0; ni < 4; ++ni) {
          // B 32x16 bf16 layout: lane half h holds K = {8h..8h+7, 16+8h..16+8h+7} of column n
          const unsigned short* r = sB + (nblk + 16 * ni + lmod) * KCP;
          v16bf bfrag = load_frag(r + kk + 8 * lhalf, r + kk + 16 + 8 * lhalf);
          #pragma unroll
          for (int mi = 0; mi < 2; ++mi)
            acc[mi][ni] = __builtin_amdgcn_wmma_f32_16x16x32_bf16(
                false, afrag[mi], false, bfrag, (short)0, acc[mi][ni],
                false, false);
        }
      }
      __syncthreads();   // after final chunk: all waves done reading sA/sB
    }

    // bias + ReLU, C-layout (lane: n=lmod, m=r+8*lhalf) -> sY
    const float* bias = bs[tw] + e * U_;
    #pragma unroll
    for (int ni = 0; ni < 4; ++ni) {
      int col = nblk + 16 * ni + lmod;
      float bv = bias[col];
      #pragma unroll
      for (int mi = 0; mi < 2; ++mi)
        #pragma unroll
        for (int r = 0; r < 8; ++r) {
          int row = mblk + 16 * mi + 8 * lhalf + r;
          float y = acc[mi][ni][r] + bv;
          sY[row * YP + col] = y > 0.f ? y : 0.f;
        }
    }
    __syncthreads();

    // weighted 8-row reduction into the 3 output accumulators
    #pragma unroll
    for (int j = 0; j < 8; ++j) {
      const float* yv = sY + (8 * p + j) * YP + u0;
      float csh, c1, c2;
      if (tw == 0)      { csh = crow[48 + j]; c1 = crow[0 + j];  c2 = crow[24 + j]; }
      else if (tw == 1) { csh = crow[40 + j]; c1 = crow[8 + j];  c2 = 0.f; }
      else              { csh = crow[56 + j]; c1 = crow[16 + j]; c2 = crow[32 + j]; }
      #pragma unroll
      for (int q = 0; q < 8; ++q) {
        float y = yv[q];
        acc_sh[q] = fmaf(csh, y, acc_sh[q]);
        acc1[q]   = fmaf(c1,  y, acc1[q]);
        acc2[q]   = fmaf(c2,  y, acc2[q]);
      }
    }
    __syncthreads();   // before next tower overwrites sY/sA/sB
  }

  // each (row, u-slab) written exactly once -> no init / atomics needed
  size_t o = (size_t)bprime * U_ + u0;
  float* o_sh = out + o;
  float* o_1  = out + (size_t)B_ * U_ + o;
  float* o_2  = out + 2 * (size_t)B_ * U_ + o;
  #pragma unroll
  for (int q = 0; q < 8; ++q) { o_sh[q] = acc_sh[q]; o_1[q] = acc1[q]; o_2[q] = acc2[q]; }
}

// ---------------------------------------------------------------------------
extern "C" void kernel_launch(void* const* d_in, const int* in_sizes, int n_in,
                              void* d_out, int out_size, void* d_ws, size_t ws_size,
                              hipStream_t stream) {
  const float* x_shared = (const float*)d_in[0];
  const float* x_task1  = (const float*)d_in[1];
  const float* x_task2  = (const float*)d_in[2];
  const float* uikey    = (const float*)d_in[3];
  const float* upkey    = (const float*)d_in[4];
  const float* ipkey    = (const float*)d_in[5];

  float* Cws = (float*)d_ws;                                        // 8 MB
  unsigned short* wbf =
      (unsigned short*)((char*)d_ws + (size_t)B_ * 64 * sizeof(float));

  const int nW = E_ * U_ * I_;                                      // 393216
  convert_w_kernel<<<(nW + 255) / 256, 256, 0, stream>>>(
      (const float*)d_in[6],  wbf + 0 * (size_t)nW, nW);            // W_exp_sh
  convert_w_kernel<<<(nW + 255) / 256, 256, 0, stream>>>(
      (const float*)d_in[8],  wbf + 1 * (size_t)nW, nW);            // W_exp_t1
  convert_w_kernel<<<(nW + 255) / 256, 256, 0, stream>>>(
      (const float*)d_in[10], wbf + 2 * (size_t)nW, nW);            // W_exp_t2

  gates_kernel<<<B_ / 256, 256, 0, stream>>>(
      x_shared, x_task1, x_task2, uikey, upkey, ipkey,
      (const float*)d_in[12], (const float*)d_in[13],   // Wg_shared, bg_shared
      (const float*)d_in[14], (const float*)d_in[15],   // Wg_t1, bg_t1
      (const float*)d_in[16], (const float*)d_in[17],   // Wg_t2, bg_t2
      (const float*)d_in[18], (const float*)d_in[19],   // Wg_share2, bg_share2
      (const float*)d_in[20], (const float*)d_in[21],   // Wg_t2_ip, bg_t2_ip
      (const float*)d_in[22], (const float*)d_in[23],   // Wg_t2_up, bg_t2_up
      (const float*)d_in[24], (const float*)d_in[25],   // Wg_s1_ip, bg_s1_ip
      (const float*)d_in[26], (const float*)d_in[27],   // Wg_s1_up, bg_s1_up
      (const float*)d_in[28], (const float*)d_in[29],   // Wg_t1_ui, bg_t1_ui
      Cws);

  // 256 tiles x 8 experts; consecutive blocks share an X tile (L2 reuse)
  expert_mix_kernel<<<(B_ / 128) * E_, 256, 0, stream>>>(
      x_shared, x_task1, x_task2, wbf,
      (const float*)d_in[7], (const float*)d_in[9], (const float*)d_in[11],
      Cws, (float*)d_out);
}
